// GaussianWrite_64201171141018
// MI455X (gfx1250) — matrix operand order repaired and checked by reading, not compiled, not run on hardware
//
#include <hip/hip_runtime.h>

// ---------------------------------------------------------------------------
// MI455X (gfx1250) persistent-tile recurrent kernel.
//
//  * grid = B/16 persistent blocks; each owns batch rows [b0, b0+16) and runs
//    all 256 steps, h kept in LDS (f32), GEMM via v_wmma_f32_16x16x32_f16
//    with f32 accumulation (A rebuilt per step in LDS as f16). The compiler
//    hoists all 16 W-fragments into VGPRs, so the K-loop is ds_load_b128 +
//    WMMA only.
//  * slot memory [B,64,256] f32 = 134 MB lives in d_ws -> resident in 192 MB
//    L2; each block touches only its own 1 MB slice (no cross-block races).
//  * pointer(t) = t % 64 is batch-uniform -> gather index + 5 Gaussian
//    softmax weights are scalar per-step values.
//  * tanh via the CDNA5 hardware V_TANH_F32 trans op when the builtin exists
//    (32 tanh/thread/step: the dominant non-WMMA VALU cost otherwise).
//  * Layernorm: per-row wave32 shfl_xor reduction.
// ---------------------------------------------------------------------------

#define HIDDEN   256
#define SLOTS    64
#define T_STEPS  256
#define TILE_B   16
#define THREADS  256
#define LN_EPS   1e-5f

typedef _Float16 f16x8  __attribute__((ext_vector_type(8)));
typedef _Float16 f16x16 __attribute__((ext_vector_type(16)));
typedef float    f32x8  __attribute__((ext_vector_type(8)));

// CDNA5 hardware tanh (TRANS32 op) with compile-safe fallback.
#if __has_builtin(__builtin_amdgcn_tanhf)
__device__ __forceinline__ float fast_tanh(float v) { return __builtin_amdgcn_tanhf(v); }
#elif __has_builtin(__builtin_amdgcn_tanh_f32)
__device__ __forceinline__ float fast_tanh(float v) { return __builtin_amdgcn_tanh_f32(v); }
#else
__device__ __forceinline__ float fast_tanh(float v) { return tanhf(v); }
#endif

// Zero the slot memory and convert W_update (f32 [n][k]) -> f16 [n][k].
__global__ void prep_kernel(const float* __restrict__ Wu,
                            _Float16* __restrict__ W16,
                            float* __restrict__ mem, long memElems) {
  long i = (long)blockIdx.x * blockDim.x + threadIdx.x;
  if (i < HIDDEN * HIDDEN) W16[i] = (_Float16)Wu[i];
  long stride = (long)gridDim.x * blockDim.x;
  for (long j = i; j < memElems; j += stride) mem[j] = 0.0f;
}

__global__ __launch_bounds__(THREADS) void rnn_kernel(
    const float* __restrict__ x,      // [B, T]
    const float* __restrict__ We,     // [256]  (W_embed is [256,1])
    const float* __restrict__ be,     // [256]
    const float* __restrict__ bu,     // [256]
    const float* __restrict__ gamma,  // [256]
    const float* __restrict__ beta,   // [256]
    const float* __restrict__ Wout,   // [10,256]
    const float* __restrict__ bout,   // [10]
    const float* __restrict__ csr,    // [1]
    const _Float16* __restrict__ W16, // [256,256] f16, row n = output channel
    float* __restrict__ mem,          // [B,64,256] f32 workspace
    float* __restrict__ out)          // [B,10]
{
  __shared__ _Float16 A_lds[TILE_B * HIDDEN];   //  8 KB: f16 A matrix (combined+h)
  __shared__ float    h_lds[TILE_B * HIDDEN];   // 16 KB: f32 hidden state
  __shared__ float    we_s[HIDDEN], be_s[HIDDEN], bu_s[HIDDEN];
  __shared__ float    g_s[HIDDEN], bt_s[HIDDEN];

  const int tid = threadIdx.x;
  const int b0  = blockIdx.x * TILE_B;

  // ---- prologue: stage parameters, zero h ----
  we_s[tid] = We[tid];  be_s[tid] = be[tid];  bu_s[tid] = bu[tid];
  g_s[tid]  = gamma[tid]; bt_s[tid] = beta[tid];
#pragma unroll
  for (int i = 0; i < (TILE_B * HIDDEN) / THREADS; ++i)
    h_lds[tid + i * THREADS] = 0.0f;
  const float cs = 1.0f / (1.0f + __expf(-csr[0]));   // sigmoid(context_strength)
  __syncthreads();

  const int lane  = tid & 31;
  const int wv    = tid >> 5;          // wave 0..7
  const int colA  = lane & 15;         // A: row m     / B: col n within tile
  const int koffA = (lane >> 4) << 4;  // A lanes 16-31 hold K 16..31
  const int koffB = (lane >> 4) << 4;  // B lanes 16-31 hold K 16..31

  for (int t = 0; t < T_STEPS; ++t) {
    // ---- scalar per-step: pointer + Gaussian weights (batch-uniform) ----
    const int base = t & (SLOTS - 1);
    int   idxs[5];
    float wgt[5];
    float wsum = 0.0f;
#pragma unroll
    for (int j = 0; j < 5; ++j) {
      int id = (base + j - 2 + SLOTS) & (SLOTS - 1);
      idxs[j] = id;
      float d = (float)id - (float)base;              // wrapped -> |d|~62 -> exp ~ 0
      float e = __expf(-(d * d) * 0.125f);
      wgt[j] = e;  wsum += e;
    }
    const float winv = 1.0f / wsum;
#pragma unroll
    for (int j = 0; j < 5; ++j) wgt[j] *= winv;

    // ---- Phase A: A = f16( tanh(x*We+be) + cs*context + h )  (16 elems/thr)
    {
      const int m  = tid >> 4;
      const int kb = (tid & 15) * 16;
      const float xv = x[(long)(b0 + m) * T_STEPS + t];
      const float* ctx = mem + ((long)(b0 + m) * SLOTS + base) * HIDDEN + kb;
      const float* hp  = &h_lds[m * HIDDEN + kb];
      _Float16*    ap  = &A_lds[m * HIDDEN + kb];
      float vals[16];
#pragma unroll
      for (int q = 0; q < 4; ++q) {
        float4 c4 = *(const float4*)(ctx + q * 4);
        float4 h4 = *(const float4*)(hp + q * 4);
        int k = kb + q * 4;
        vals[q*4+0] = fast_tanh(xv * we_s[k+0] + be_s[k+0]) + cs * c4.x + h4.x;
        vals[q*4+1] = fast_tanh(xv * we_s[k+1] + be_s[k+1]) + cs * c4.y + h4.y;
        vals[q*4+2] = fast_tanh(xv * we_s[k+2] + be_s[k+2]) + cs * c4.z + h4.z;
        vals[q*4+3] = fast_tanh(xv * we_s[k+3] + be_s[k+3]) + cs * c4.w + h4.w;
      }
      f16x8 pk0, pk1;
#pragma unroll
      for (int i = 0; i < 8; ++i) { pk0[i] = (_Float16)vals[i]; pk1[i] = (_Float16)vals[8 + i]; }
      *(f16x8*)(ap)     = pk0;
      *(f16x8*)(ap + 8) = pk1;
      if (t + 1 < T_STEPS)   // prefetch next step's context slot into near caches
        __builtin_prefetch(mem + ((long)(b0 + m) * SLOTS + ((base + 1) & (SLOTS - 1))) * HIDDEN + kb, 0, 3);
    }
    __syncthreads();

    // ---- Phase B: D = A(16x256) x W^T via 16x16x32 f16 WMMA, f32 accum ----
    f32x8 acc0 = {};
    f32x8 acc1 = {};
    {
      const _Float16* Arow  = &A_lds[colA * HIDDEN];
      const _Float16* Brow0 = W16 + (long)((wv * 2 + 0) * 16 + colA) * HIDDEN;
      const _Float16* Brow1 = W16 + (long)((wv * 2 + 1) * 16 + colA) * HIDDEN;
#pragma unroll
      for (int kt = 0; kt < 8; ++kt) {
        const int kb = kt * 32;
        // A fragment: lane<16 K=0..15, lane>=16 K=16..31 (contiguous)
        f16x8 alo = *(const f16x8*)(Arow + kb + koffA);
        f16x8 ahi = *(const f16x8*)(Arow + kb + koffA + 8);
        f16x16 a = __builtin_shufflevector(alo, ahi, 0,1,2,3,4,5,6,7,8,9,10,11,12,13,14,15);
        // B fragment: lane<16 K=0..15, lane>=16 K=16..31 (contiguous)
        f16x8 b0l = *(const f16x8*)(Brow0 + kb + koffB);
        f16x8 b0h = *(const f16x8*)(Brow0 + kb + koffB + 8);
        f16x16 bb0 = __builtin_shufflevector(b0l, b0h, 0,1,2,3,4,5,6,7,8,9,10,11,12,13,14,15);
        acc0 = __builtin_amdgcn_wmma_f32_16x16x32_f16(false, a, false, bb0,
                                                      (short)0, acc0, false, false);
        f16x8 b1l = *(const f16x8*)(Brow1 + kb + koffB);
        f16x8 b1h = *(const f16x8*)(Brow1 + kb + koffB + 8);
        f16x16 bb1 = __builtin_shufflevector(b1l, b1h, 0,1,2,3,4,5,6,7,8,9,10,11,12,13,14,15);
        acc1 = __builtin_amdgcn_wmma_f32_16x16x32_f16(false, a, false, bb1,
                                                      (short)0, acc1, false, false);
      }
    }

    // ---- Phase C: h_pre = tanh(D + b_update) -> h_lds (C/D layout scatter)
    {
      const int rb = (lane >> 4) * 8;       // M = rb + r
      const int n0 = (wv * 2 + 0) * 16 + colA;
      const int n1 = n0 + 16;
      const float bu0 = bu_s[n0], bu1 = bu_s[n1];
#pragma unroll
      for (int r = 0; r < 8; ++r) {
        h_lds[(rb + r) * HIDDEN + n0] = fast_tanh(acc0[r] + bu0);
        h_lds[(rb + r) * HIDDEN + n1] = fast_tanh(acc1[r] + bu1);
      }
    }
    __syncthreads();

    // ---- Phase D: layernorm per row (wave32 reduce) + 5-slot scatter-add --
#pragma unroll
    for (int rr = 0; rr < 2; ++rr) {
      const int m = wv * 2 + rr;
      float* hr = &h_lds[m * HIDDEN];
      const int n0 = lane * 8;
      float4 a4 = *(float4*)(hr + n0);
      float4 b4 = *(float4*)(hr + n0 + 4);
      float v[8] = { a4.x, a4.y, a4.z, a4.w, b4.x, b4.y, b4.z, b4.w };
      float s = 0.0f, s2 = 0.0f;
#pragma unroll
      for (int i = 0; i < 8; ++i) { s += v[i]; s2 += v[i] * v[i]; }
#pragma unroll
      for (int msk = 16; msk >= 1; msk >>= 1) {
        s  += __shfl_xor(s,  msk, 32);
        s2 += __shfl_xor(s2, msk, 32);
      }
      const float mu   = s * (1.0f / HIDDEN);
      const float var  = s2 * (1.0f / HIDDEN) - mu * mu;
      const float rstd = rsqrtf(var + LN_EPS);
      float hn[8];
#pragma unroll
      for (int i = 0; i < 8; ++i) {
        const int n = n0 + i;
        hn[i] = (v[i] - mu) * rstd * g_s[n] + bt_s[n];
      }
      *(float4*)(hr + n0)     = make_float4(hn[0], hn[1], hn[2], hn[3]);
      *(float4*)(hr + n0 + 4) = make_float4(hn[4], hn[5], hn[6], hn[7]);
      // scatter-add into 5 neighboring slots (exclusive per-block region)
#pragma unroll
      for (int j = 0; j < 5; ++j) {
        float* mr = mem + ((long)(b0 + m) * SLOTS + idxs[j]) * HIDDEN + n0;
        float4 p0 = *(float4*)mr;
        float4 p1 = *(float4*)(mr + 4);
        const float wj = wgt[j];
        p0.x += wj * hn[0]; p0.y += wj * hn[1]; p0.z += wj * hn[2]; p0.w += wj * hn[3];
        p1.x += wj * hn[4]; p1.y += wj * hn[5]; p1.z += wj * hn[6]; p1.w += wj * hn[7];
        *(float4*)mr       = p0;
        *(float4*)(mr + 4) = p1;
      }
    }
    __threadfence_block();   // slot-memory stores visible before next gather
    __syncthreads();
  }

  // ---- epilogue: out = h @ W_out^T + b_out  (160 dot products) ----
  if (tid < TILE_B * 10) {
    const int m = tid / 10, o = tid % 10;
    const float* wr = Wout + o * HIDDEN;
    const float* hr = &h_lds[m * HIDDEN];
    float sum = 0.0f;
#pragma unroll 8
    for (int k = 0; k < HIDDEN; ++k) sum += hr[k] * wr[k];
    out[(long)(b0 + m) * 10 + o] = sum + bout[o];
  }
}

extern "C" void kernel_launch(void* const* d_in, const int* in_sizes, int n_in,
                              void* d_out, int out_size, void* d_ws, size_t ws_size,
                              hipStream_t stream) {
  const float* x     = (const float*)d_in[0];
  const float* We    = (const float*)d_in[1];
  const float* be    = (const float*)d_in[2];
  const float* Wu    = (const float*)d_in[3];
  const float* bu    = (const float*)d_in[4];
  const float* gamma = (const float*)d_in[5];
  const float* beta  = (const float*)d_in[6];
  const float* Wout  = (const float*)d_in[7];
  const float* bout  = (const float*)d_in[8];
  const float* csr   = (const float*)d_in[9];

  const int B = in_sizes[0] / T_STEPS;          // 2048
  const long memElems = (long)B * SLOTS * HIDDEN;

  // workspace: [0,256KB) f16 W_update ; [256KB, ...) f32 slot memory (134 MB)
  _Float16* W16 = (_Float16*)d_ws;
  float*    mem = (float*)((char*)d_ws + (256 << 10));

  prep_kernel<<<1024, 256, 0, stream>>>(Wu, W16, mem, memElems);
  rnn_kernel<<<B / TILE_B, THREADS, 0, stream>>>(
      x, We, be, bu, gamma, beta, Wout, bout, csr, W16, mem, (float*)d_out);
}
